// CausalAttention_5841155523204
// MI455X (gfx1250) — matrix operand order repaired and checked by reading, not compiled, not run on hardware
//
#include <hip/hip_runtime.h>
#include <hip/hip_bf16.h>
#include <math.h>

// ---------- types ----------
typedef __attribute__((ext_vector_type(16))) __bf16 v16bf;
typedef __attribute__((ext_vector_type(8)))  __bf16 v8bf;
typedef __attribute__((ext_vector_type(4)))  __bf16 v4bf;
typedef __attribute__((ext_vector_type(8)))  float  v8f;
typedef __attribute__((ext_vector_type(4)))  int    v4i;

__device__ __forceinline__ __bf16 f2bf(float f) { return (__bf16)f; }

// ---------- async global->LDS path (gfx1250), with safe fallback ----------
#if defined(__gfx1250__) && __has_builtin(__builtin_amdgcn_global_load_async_to_lds_b128)
#define USE_ASYNC_LDS 1
typedef __attribute__((address_space(1))) v4i gv4i;   // global int4
typedef __attribute__((address_space(3))) v4i lv4i;   // LDS int4
__device__ __forceinline__ void async_ld16(const __bf16* g, __bf16* l) {
    __builtin_amdgcn_global_load_async_to_lds_b128((gv4i*)g, (lv4i*)l, 0, 0);
}
__device__ __forceinline__ void wait_async0() {
#if __has_builtin(__builtin_amdgcn_s_wait_asynccnt)
    __builtin_amdgcn_s_wait_asynccnt(0);
#else
    asm volatile("s_wait_asynccnt 0x0" ::: "memory");
#endif
}
#else
#define USE_ASYNC_LDS 0
#endif

// ---------- problem constants ----------
#define BATCH 4
#define TSEQ  2048
#define DMODEL 1024

// ---------- GEMM tiling ----------
#define BLOCK_M 128   // 4 wave-rows x 32
#define BLOCK_N 128   // 2 wave-cols x 64
#define KSTEP   32
#define LDSTR   40    // bf16 elems per LDS row (32 + 8 pad) -> 80B, 16B aligned

// ============================================================
// f32 -> bf16 convert (vectorized by 4)
// ============================================================
__global__ __launch_bounds__(256)
void qkv_f32_to_bf16(const float* __restrict__ in, __bf16* __restrict__ out, int n4) {
    int i = blockIdx.x * 256 + threadIdx.x;
    if (i >= n4) return;
    const float4 v = ((const float4*)in)[i];
    v4bf o;
    o[0] = f2bf(v.x); o[1] = f2bf(v.y); o[2] = f2bf(v.z); o[3] = f2bf(v.w);
    ((v4bf*)out)[i] = o;
}

// ============================================================
// Weight transpose + convert: Wt[o][i] = (bf16)W[i][o]
// ============================================================
__global__ __launch_bounds__(256)
void qkv_transpose_w(const float* __restrict__ W, __bf16* __restrict__ Wt, int Din, int Dout) {
    int i = blockIdx.x * 256 + threadIdx.x;            // index into Wt, row-major [Dout][Din]
    if (i >= Din * Dout) return;
    int o  = i / Din;
    int ii = i - o * Din;
    Wt[i] = f2bf(W[(size_t)ii * Dout + o]);
}

// ============================================================
// Generic bf16 WMMA GEMM: C[M x N] = A[M x K] * Bt[N x K]^T
//   A, Bt: bf16 row-major over K (ldA / ldB in elements)
// 8 waves arranged 4(M) x 2(N); each wave computes 32x64 (8 wmma accs).
// Async path: double-buffered LDS, loads for tile k+1 issued while
// computing tile k, ONE barrier per K-step.
// Epilogue modes:
//   0: bf16 out, val*scale                  (Q, scale = 1/sqrt(D))
//   1: bf16 out                             (K)
//   2: bf16 out transposed per batch: Out[(b*outStride + gN)*Ttok + t]  (V^T)
//   3: f32 out, causal mask (gN<=gM else -inf)                          (scores)
//   4: f32 out                              (attn output)
// ============================================================
__global__ __launch_bounds__(256)
void wmma_gemm_bf16(const __bf16* __restrict__ A, const __bf16* __restrict__ Bt,
                    void* __restrict__ Out,
                    int K, int ldA, int ldB, int outStride,
                    int mode, int Ttok, float scale) {
#if USE_ASYNC_LDS
    __shared__ __bf16 As[2][BLOCK_M * LDSTR];
    __shared__ __bf16 Bs[2][BLOCK_N * LDSTR];
#else
    __shared__ __bf16 As[1][BLOCK_M * LDSTR];
    __shared__ __bf16 Bs[1][BLOCK_N * LDSTR];
#endif

    const int tid    = threadIdx.x;
    const int lane   = tid & 31;
    const int wave   = tid >> 5;                // 0..7
    const int waveM  = (wave & 3) * 32;         // wave row origin in block tile
    const int waveN  = (wave >> 2) * 64;        // wave col origin in block tile
    const int blockM = blockIdx.x * BLOCK_M;
    const int blockN = blockIdx.y * BLOCK_N;

    // Fully-masked score blocks: vectorized -inf fill, skip the GEMM.
    if (mode == 3 && blockN > blockM + (BLOCK_M - 1)) {
        float* o = (float*)Out;
        const float4 ninf = make_float4(-INFINITY, -INFINITY, -INFINITY, -INFINITY);
        for (int i = tid; i < (BLOCK_M * BLOCK_N) / 4; i += 256) {
            int r = i >> 5, c = (i & 31) * 4;
            *(float4*)&o[(size_t)(blockM + r) * outStride + (blockN + c)] = ninf;
        }
        return;
    }

    v8f acc[2][4] = {};   // [m-tile][n-tile] 16x16 f32 each

    const int laneHalf = lane >> 4;     // 0 or 1
    const int laneMod  = lane & 15;

    // per-thread staging coordinates: 32B of one A row + 32B of one B row
    const int sr = tid >> 1, sh = tid & 1;
    const __bf16* srcA0 = A  + (size_t)(blockM + sr) * ldA + sh * 16;
    const __bf16* srcB0 = Bt + (size_t)(blockN + sr) * ldB + sh * 16;
    const int sOff = sr * LDSTR + sh * 16;

    // compute on one LDS buffer
    auto compute = [&](const __bf16* asb, const __bf16* bsb) {
        // A fragments (16x32, ISA layout): lane l, row = l%16,
        // elems 0..7 -> K=(l/16)*8+j ; elems 8..15 -> K=16+(l/16)*8+(j-8)
        v16bf afrag[2];
        #pragma unroll
        for (int mt = 0; mt < 2; ++mt) {
            const int arow = waveM + mt * 16 + laneMod;
            v8bf alo = *(const v8bf*)(&asb[arow * LDSTR + laneHalf * 8]);
            v8bf ahi = *(const v8bf*)(&asb[arow * LDSTR + 16 + laneHalf * 8]);
            afrag[mt] = __builtin_shufflevector(alo, ahi,
                0, 1, 2, 3, 4, 5, 6, 7, 8, 9, 10, 11, 12, 13, 14, 15);
        }
        // B fragments (32x16): lane l, col = l%16, elems 0..7 -> K=(l/16)*8+j ; elems 8..15 -> K=16+(l/16)*8+(j-8)
        #pragma unroll
        for (int nt = 0; nt < 4; ++nt) {
            const int brow = waveN + nt * 16 + laneMod;
            v8bf blo = *(const v8bf*)(&bsb[brow * LDSTR + laneHalf * 8]);
            v8bf bhi = *(const v8bf*)(&bsb[brow * LDSTR + 16 + laneHalf * 8]);
            v16bf bfrag = __builtin_shufflevector(blo, bhi,
                0, 1, 2, 3, 4, 5, 6, 7, 8, 9, 10, 11, 12, 13, 14, 15);
            #pragma unroll
            for (int mt = 0; mt < 2; ++mt) {
                acc[mt][nt] = __builtin_amdgcn_wmma_f32_16x16x32_bf16(
                    false, afrag[mt], false, bfrag, (short)0, acc[mt][nt], false, false);
            }
        }
    };

#if USE_ASYNC_LDS
    // stage tile at k-offset into buffer buf (4 async b128 per thread)
    auto stage = [&](int k0, int buf) {
        const __bf16* sa = srcA0 + k0;
        const __bf16* sb = srcB0 + k0;
        __bf16* da = &As[buf][sOff];
        __bf16* db = &Bs[buf][sOff];
        async_ld16(sa,     da);
        async_ld16(sa + 8, da + 8);
        async_ld16(sb,     db);
        async_ld16(sb + 8, db + 8);
    };

    stage(0, 0);                       // prologue
    int kb = 0;
    for (int k0 = 0; k0 < K; k0 += KSTEP) {
        wait_async0();                 // tile k resident (this wave's loads)
        __syncthreads();               // all waves' loads resident; prev reads of other buf done
        if (k0 + KSTEP < K) {
            stage(k0 + KSTEP, kb ^ 1); // overlap next tile's loads with compute
            __builtin_prefetch(srcA0 + k0 + 2 * KSTEP, 0, 1);
            __builtin_prefetch(srcB0 + k0 + 2 * KSTEP, 0, 1);
        }
        compute(As[kb], Bs[kb]);
        kb ^= 1;
    }
#else
    for (int k0 = 0; k0 < K; k0 += KSTEP) {
        v8bf a0 = *(const v8bf*)(srcA0 + k0);
        v8bf a1 = *(const v8bf*)(srcA0 + k0 + 8);
        v8bf b0 = *(const v8bf*)(srcB0 + k0);
        v8bf b1 = *(const v8bf*)(srcB0 + k0 + 8);
        *(v8bf*)(&As[0][sOff])     = a0;
        *(v8bf*)(&As[0][sOff + 8]) = a1;
        *(v8bf*)(&Bs[0][sOff])     = b0;
        *(v8bf*)(&Bs[0][sOff + 8]) = b1;
        __syncthreads();
        compute(As[0], Bs[0]);
        __syncthreads();
    }
#endif

    // ---- epilogue: C element r -> M = r + (l/16)*8, N = l%16 ----
    #pragma unroll
    for (int mt = 0; mt < 2; ++mt) {
        const int mBase = blockM + waveM + mt * 16 + laneHalf * 8;
        #pragma unroll
        for (int nt = 0; nt < 4; ++nt) {
            const int gN = blockN + waveN + nt * 16 + laneMod;
            #pragma unroll
            for (int r = 0; r < 8; ++r) {
                const float v = acc[mt][nt][r];
                const int gM = mBase + r;
                if (mode == 0) {
                    ((__bf16*)Out)[(size_t)gM * outStride + gN] = f2bf(v * scale);
                } else if (mode == 1) {
                    ((__bf16*)Out)[(size_t)gM * outStride + gN] = f2bf(v);
                } else if (mode == 2) {
                    int b = gM / Ttok, t = gM - b * Ttok;
                    ((__bf16*)Out)[((size_t)b * outStride + gN) * Ttok + t] = f2bf(v);
                } else if (mode == 3) {
                    ((float*)Out)[(size_t)gM * outStride + gN] = (gN <= gM) ? v : -INFINITY;
                } else {
                    ((float*)Out)[(size_t)gM * outStride + gN] = v;
                }
            }
        }
    }
}

// ============================================================
// Row softmax: S (f32, T cols) -> P (bf16).  One block per row.
// ============================================================
__global__ __launch_bounds__(256)
void causal_softmax_row(const float* __restrict__ S, __bf16* __restrict__ P, int T) {
    const int row = blockIdx.x;
    const float* s = S + (size_t)row * T;
    __bf16* p      = P + (size_t)row * T;
    __shared__ float red[256];

    float lmax = -INFINITY;
    for (int c = threadIdx.x; c < T; c += 256) lmax = fmaxf(lmax, s[c]);
    red[threadIdx.x] = lmax; __syncthreads();
    for (int off = 128; off > 0; off >>= 1) {
        if (threadIdx.x < off) red[threadIdx.x] = fmaxf(red[threadIdx.x], red[threadIdx.x + off]);
        __syncthreads();
    }
    const float m = red[0];
    __syncthreads();

    float lsum = 0.0f;
    for (int c = threadIdx.x; c < T; c += 256) lsum += __expf(s[c] - m);
    red[threadIdx.x] = lsum; __syncthreads();
    for (int off = 128; off > 0; off >>= 1) {
        if (threadIdx.x < off) red[threadIdx.x] += red[threadIdx.x + off];
        __syncthreads();
    }
    const float inv = 1.0f / red[0];

    for (int c = threadIdx.x; c < T; c += 256)
        p[c] = f2bf(__expf(s[c] - m) * inv);
}

// ============================================================
// host-side launcher
// ============================================================
extern "C" void kernel_launch(void* const* d_in, const int* in_sizes, int n_in,
                              void* d_out, int out_size, void* d_ws, size_t ws_size,
                              hipStream_t stream) {
    const float* x  = (const float*)d_in[0];
    const float* Wq = (const float*)d_in[1];
    const float* Wk = (const float*)d_in[2];
    const float* Wv = (const float*)d_in[3];
    float* out = (float*)d_out;

    const int B = BATCH, T = TSEQ, D = DMODEL;
    const float scale = 0.03125f;   // 1/sqrt(1024)

    // ---- workspace layout (bytes) ----
    char* ws = (char*)d_ws;
    size_t off = 0;
    __bf16* xb  = (__bf16*)(ws + off); off += (size_t)B * T * D * 2;   // x bf16
    __bf16* Wqt = (__bf16*)(ws + off); off += (size_t)D * D * 2;       // Wq^T bf16 [out][in]
    __bf16* Wkt = (__bf16*)(ws + off); off += (size_t)D * D * 2;
    __bf16* Wvt = (__bf16*)(ws + off); off += (size_t)D * D * 2;
    __bf16* Qb  = (__bf16*)(ws + off); off += (size_t)B * T * D * 2;   // Q (pre-scaled)
    __bf16* Kb  = (__bf16*)(ws + off); off += (size_t)B * T * D * 2;   // K
    __bf16* Vt  = (__bf16*)(ws + off); off += (size_t)B * T * D * 2;   // V^T  [b][d][t]
    float*  S   = (float*)(ws + off);  off += (size_t)T * T * 4;       // per-batch scores
    __bf16* Pb  = (__bf16*)(ws + off); off += (size_t)T * T * 2;       // per-batch probs

    dim3 blk(256);

    // ---- 1. conversions ----
    int n4 = (B * T * D) / 4;
    qkv_f32_to_bf16<<<dim3((n4 + 255) / 256), blk, 0, stream>>>(x, xb, n4);
    int nw = D * D;
    qkv_transpose_w<<<dim3((nw + 255) / 256), blk, 0, stream>>>(Wq, Wqt, D, D);
    qkv_transpose_w<<<dim3((nw + 255) / 256), blk, 0, stream>>>(Wk, Wkt, D, D);
    qkv_transpose_w<<<dim3((nw + 255) / 256), blk, 0, stream>>>(Wv, Wvt, D, D);

    // ---- 2. QKV projections:  [B*T x D] = xb * W^T ----
    dim3 gQKV((B * T) / BLOCK_M, D / BLOCK_N);
    wmma_gemm_bf16<<<gQKV, blk, 0, stream>>>(xb, Wqt, Qb, D, D, D, D, 0, T, scale);
    wmma_gemm_bf16<<<gQKV, blk, 0, stream>>>(xb, Wkt, Kb, D, D, D, D, 1, T, 1.0f);
    wmma_gemm_bf16<<<gQKV, blk, 0, stream>>>(xb, Wvt, Vt, D, D, D, D, 2, T, 1.0f);

    // ---- 3. attention, one batch at a time (stream-ordered reuse of S/Pb) ----
    for (int b = 0; b < B; ++b) {
        const __bf16* Qbb = Qb + (size_t)b * T * D;
        const __bf16* Kbb = Kb + (size_t)b * T * D;
        const __bf16* Vtb = Vt + (size_t)b * D * T;
        float* outb = out + (size_t)b * T * D;

        dim3 gS(T / BLOCK_M, T / BLOCK_N);
        wmma_gemm_bf16<<<gS, blk, 0, stream>>>(Qbb, Kbb, S, D, D, D, T, 3, T, 1.0f);

        causal_softmax_row<<<dim3(T), blk, 0, stream>>>(S, Pb, T);

        dim3 gO(T / BLOCK_M, D / BLOCK_N);
        wmma_gemm_bf16<<<gO, blk, 0, stream>>>(Pb, Vtb, outb, T, T, T, D, 4, T, 1.0f);
    }
}
